// Net_13451837571224
// MI455X (gfx1250) — matrix operand order repaired and checked by reading, not compiled, not run on hardware
//
#include <hip/hip_runtime.h>
#include <hip/hip_bf16.h>

// ---------------------------------------------------------------------------
// GCN (3 layers) + Graph Multiset Transformer pooling for MI455X (gfx1250).
// Dense projections -> v_wmma_f32_16x16x32_f16 (f16 in, f32 accumulate),
// one wave per 16-row strip computing the full output width. All weight
// (B) fragments are preloaded into registers before the K loop so the
// WMMA sequence is not serialized behind per-tile load drains.
// Edge aggregation is the memory-bound part: global_atomic_add_f32 scatter.
// ---------------------------------------------------------------------------

typedef _Float16 h16;
typedef __attribute__((ext_vector_type(16))) _Float16 v16h;
typedef __attribute__((ext_vector_type(8)))  _Float16 v8h;
typedef __attribute__((ext_vector_type(8)))  float    v8f;

#define N_NODES 100000
#define N_EDGES 3200000
#define NGRAPH  250
#define NPER    400
#define H1      32
#define CCH     96
#define NHEADS  4
#define DHEAD   24

// ---------------- elementwise / graph kernels ----------------

__global__ void k_set1(float* __restrict__ p, int n) {
    int t = blockIdx.x * blockDim.x + threadIdx.x;
    if (t < n) p[t] = 1.0f;
}

__global__ void k_zero(float* __restrict__ p, int n) {
    int t = blockIdx.x * blockDim.x + threadIdx.x;
    if (t < n) p[t] = 0.0f;
}

__global__ void k_rsqrt_ip(float* __restrict__ p, int n) {
    int t = blockIdx.x * blockDim.x + threadIdx.x;
    if (t < n) p[t] = rsqrtf(p[t]);
}

__global__ void k_degacc(const int* __restrict__ dst, float* __restrict__ deg, int e) {
    int t = blockIdx.x * blockDim.x + threadIdx.x;
    if (t < e) atomicAdd(&deg[dst[t]], 1.0f);
}

// conv1: h = x @ W1   (x: [N,3], W1: [3,32])  K=3 -> plain VALU
__global__ void k_conv1(const float* __restrict__ x, const float* __restrict__ W,
                        float* __restrict__ h) {
    int t = blockIdx.x * blockDim.x + threadIdx.x;
    if (t >= N_NODES * H1) return;
    int i = t >> 5, f = t & 31;
    const float* xr = x + i * 3;
    h[t] = xr[0] * W[f] + xr[1] * W[H1 + f] + xr[2] * W[2 * H1 + f];
}

// agg[dst] += h[src] * dinv[src]*dinv[dst]   (thread = edge*32 + feature)
__global__ void k_scatter(const int* __restrict__ src, const int* __restrict__ dst,
                          const float* __restrict__ dinv, const float* __restrict__ h,
                          float* __restrict__ agg) {
    int t = blockIdx.x * blockDim.x + threadIdx.x;
    if (t >= N_EDGES * 32) return;
    int e = t >> 5, f = t & 31;
    int s = src[e], d = dst[e];
    atomicAdd(&agg[d * H1 + f], h[s * H1 + f] * dinv[s] * dinv[d]);
}

// x_next = relu(agg + h*dinv^2 + b); write f16 next-layer input + f16 concat slice
__global__ void k_finalize(const float* __restrict__ agg, const float* __restrict__ h,
                           const float* __restrict__ dinv, const float* __restrict__ b,
                           h16* __restrict__ xnext, h16* __restrict__ Hc, int off) {
    int t = blockIdx.x * blockDim.x + threadIdx.x;
    if (t >= N_NODES * H1) return;
    int i = t >> 5, f = t & 31;
    float di = dinv[i];
    float v = agg[t] + h[t] * di * di + b[f];
    v = fmaxf(v, 0.0f);
    xnext[t] = (h16)v;
    Hc[(long long)i * CCH + off + f] = (h16)v;
}

// W[K,Nout] f32 -> WT[Nout,K] f16 (transposed so WMMA B-fragments load contiguously)
__global__ void k_prep_wt(const float* __restrict__ W, h16* __restrict__ WT,
                          int K, int Nout) {
    int t = blockIdx.x * blockDim.x + threadIdx.x;
    if (t >= K * Nout) return;
    int k = t / Nout, n = t - k * Nout;
    WT[(long long)n * K + k] = (h16)W[t];
}

__global__ void k_f2h(const float* __restrict__ s, h16* __restrict__ d, int n) {
    int t = blockIdx.x * blockDim.x + threadIdx.x;
    if (t < n) d[t] = (h16)s[t];
}

// o = a + res[i % resMod]  (broadcast residual, e.g. PMA seed)
__global__ void k_addres(const float* __restrict__ a, const float* __restrict__ res,
                         int resMod, float* __restrict__ o32, h16* __restrict__ o16, int n) {
    int t = blockIdx.x * blockDim.x + threadIdx.x;
    if (t >= n) return;
    float v = a[t] + res[t % resMod];
    o32[t] = v;
    o16[t] = (h16)v;
}

// o = a + b
__global__ void k_add2(const float* __restrict__ a, const float* __restrict__ b,
                       float* __restrict__ o32, h16* __restrict__ o16, int n) {
    int t = blockIdx.x * blockDim.x + threadIdx.x;
    if (t >= n) return;
    float v = a[t] + b[t];
    o32[t] = v;
    o16[t] = (h16)v;
}

// final classifier: out[b,c] = g[b,:16] @ W2[16,2] + b2
__global__ void k_lin2(const float* __restrict__ g, const float* __restrict__ W,
                       const float* __restrict__ b, float* __restrict__ out) {
    int t = blockIdx.x * blockDim.x + threadIdx.x;
    if (t >= NGRAPH * 2) return;
    int bi = t >> 1, c = t & 1;
    float acc = b[c];
    #pragma unroll
    for (int j = 0; j < 16; ++j) acc += g[bi * 16 + j] * W[j * 2 + c];
    out[t] = acc;
}

// ------- WMMA GEMM:  D = act(A[M,KC*32] * WT[NT*16,KC*32]^T + bias) -------
// One wave (32 threads) per 16-row strip; computes all NT column tiles.
// All NT*KC weight fragments are preloaded before the (fully unrolled) WMMA
// sequence; A-fragments are loaded once per K-chunk and reused NT times.
// A-fragment layout (16-bit A 16x32): lanes 0-15 rows, halves 0..15=K(ks..ks+15),
// ks = (lane>>4)*16.
// B-fragment: lane col = lane&15, halves = K(ks..ks+15), ks = (lane>>4)*16.
// C (f32 8 VGPRs): rows mbase..mbase+7, mbase = (lane>>4)*8, col = lane&15.
template<int NT, int KC>
__global__ void k_gemm_wmma(const h16* __restrict__ A, const h16* __restrict__ WT,
                            const float* __restrict__ bias,
                            float* __restrict__ out32, h16* __restrict__ out16,
                            int M, int relu) {
    const int K    = KC * 32;
    const int Nout = NT * 16;
    int row0 = blockIdx.x << 4;
    int lane = threadIdx.x;
    int half = lane >> 4;
    int l15  = lane & 15;
    int rA = row0 + l15;
    if (rA > M - 1) rA = M - 1;          // clamp; bad rows masked at store
    int kb = half << 3;
    int ks = half << 4;
    const h16* Arow = A + (long long)rA * K;
    const h16* Brow = WT + (long long)l15 * K;   // + nt*16*K per column tile

    // preload all weight fragments (loop-invariant for the wave)
    v16h bfrag[NT * KC];
    #pragma unroll
    for (int nt = 0; nt < NT; ++nt)
        #pragma unroll
        for (int kc = 0; kc < KC; ++kc)
            bfrag[nt * KC + kc] =
                *(const v16h*)(Brow + (long long)nt * 16 * K + kc * 32 + ks);

    v8f zero = {};
    v8f acc[NT];
    #pragma unroll
    for (int i = 0; i < NT; ++i) acc[i] = zero;

    #pragma unroll
    for (int kc = 0; kc < KC; ++kc) {
        v8h alo = *(const v8h*)(Arow + kc * 32 + ks);
        v8h ahi = *(const v8h*)(Arow + kc * 32 + ks + 8);
        v16h a = __builtin_shufflevector(alo, ahi,
                 0, 1, 2, 3, 4, 5, 6, 7, 8, 9, 10, 11, 12, 13, 14, 15);
        #pragma unroll
        for (int nt = 0; nt < NT; ++nt)
            acc[nt] = __builtin_amdgcn_wmma_f32_16x16x32_f16(
                          false, a, false, bfrag[nt * KC + kc], (short)0,
                          acc[nt], false, false);
    }

    int mbase = row0 + (half << 3);
    #pragma unroll
    for (int nt = 0; nt < NT; ++nt) {
        int cB = nt * 16 + l15;
        float bv = bias ? bias[cB] : 0.0f;
        #pragma unroll
        for (int v = 0; v < 8; ++v) {
            int m = mbase + v;
            if (m < M) {
                float val = acc[nt][v] + bv;
                if (relu) val = fmaxf(val, 0.0f);
                long long o = (long long)m * Nout + cB;
                if (out32) out32[o] = val;
                if (out16) out16[o] = (h16)val;
            }
        }
    }
}

// ---------------- per-graph multihead attention (4 heads, d=24, C=96) ----------------
// scores kept in dynamic LDS: NHEADS*Lq*Lk floats (max 64000 B for Lq=10,Lk=400)
__global__ void k_attn(const h16* __restrict__ Q, int qStride,
                       const h16* __restrict__ K, const h16* __restrict__ V,
                       h16* __restrict__ out, int Lq, int Lk) {
    extern __shared__ float sc[];
    int g = blockIdx.x;
    const h16* Qg = Q + (long long)g * qStride;
    const h16* Kg = K + (long long)g * Lk * CCH;
    const h16* Vg = V + (long long)g * Lk * CCH;

    int total = NHEADS * Lq * Lk;
    for (int idx = threadIdx.x; idx < total; idx += blockDim.x) {
        int k = idx % Lk;
        int r = idx / Lk;            // r = h*Lq + q
        int q = r % Lq;
        int h = r / Lq;
        const h16* qp = Qg + q * CCH + h * DHEAD;
        const h16* kp = Kg + (long long)k * CCH + h * DHEAD;
        float acc = 0.0f;
        #pragma unroll
        for (int j = 0; j < DHEAD; ++j) acc += (float)qp[j] * (float)kp[j];
        sc[idx] = acc * 0.20412414523193154f;   // 1/sqrt(24)
    }
    __syncthreads();

    int rows = NHEADS * Lq;
    for (int r = threadIdx.x; r < rows; r += blockDim.x) {
        float* row = sc + r * Lk;
        float m = -3.0e38f;
        for (int k = 0; k < Lk; ++k) m = fmaxf(m, row[k]);
        float s = 0.0f;
        for (int k = 0; k < Lk; ++k) { float e = __expf(row[k] - m); row[k] = e; s += e; }
        float inv = 1.0f / s;
        for (int k = 0; k < Lk; ++k) row[k] *= inv;
    }
    __syncthreads();

    int ot = Lq * CCH;
    for (int idx = threadIdx.x; idx < ot; idx += blockDim.x) {
        int c = idx % CCH;
        int q = idx / CCH;
        int h = c / DHEAD;
        const float* row = sc + (h * Lq + q) * Lk;
        float acc = 0.0f;
        for (int k = 0; k < Lk; ++k) acc += row[k] * (float)Vg[(long long)k * CCH + c];
        out[((long long)g * Lq + q) * CCH + c] = (h16)acc;
    }
}

// ---------------------------------------------------------------------------

extern "C" void kernel_launch(void* const* d_in, const int* in_sizes, int n_in,
                              void* d_out, int out_size, void* d_ws, size_t ws_size,
                              hipStream_t stream) {
    (void)in_sizes; (void)n_in; (void)out_size; (void)ws_size;

    // ---- inputs (insertion-order flatten of setup_inputs dict) ----
    const float* x    = (const float*)d_in[0];
    const int*   esrc = (const int*)d_in[1];
    const int*   edst = esrc + N_EDGES;
    // d_in[2] = batch (arange//400 -> pure reshape, unused)
    const float* c1W = (const float*)d_in[3];  const float* c1b = (const float*)d_in[4];
    const float* c2W = (const float*)d_in[5];  const float* c2b = (const float*)d_in[6];
    const float* c3W = (const float*)d_in[7];  const float* c3b = (const float*)d_in[8];
    const float* p1linW = (const float*)d_in[9];  const float* p1linb = (const float*)d_in[10];
    const float* p1seed = (const float*)d_in[11];
    const float* p1Wq = (const float*)d_in[12]; const float* p1bq = (const float*)d_in[13];
    const float* p1Wk = (const float*)d_in[14]; const float* p1bk = (const float*)d_in[15];
    const float* p1Wv = (const float*)d_in[16]; const float* p1bv = (const float*)d_in[17];
    const float* p1Wo = (const float*)d_in[18]; const float* p1bo = (const float*)d_in[19];
    const float* p1fW = (const float*)d_in[20]; const float* p1fb = (const float*)d_in[21];
    const float* sWq = (const float*)d_in[22];  const float* sbq = (const float*)d_in[23];
    const float* sWk = (const float*)d_in[24];  const float* sbk = (const float*)d_in[25];
    const float* sWv = (const float*)d_in[26];  const float* sbv = (const float*)d_in[27];
    const float* sWo = (const float*)d_in[28];  const float* sbo = (const float*)d_in[29];
    const float* sfW = (const float*)d_in[30];  const float* sfb = (const float*)d_in[31];
    const float* p2linW = (const float*)d_in[32]; const float* p2linb = (const float*)d_in[33];
    const float* p2seed = (const float*)d_in[34];
    const float* p2Wq = (const float*)d_in[35]; const float* p2bq = (const float*)d_in[36];
    const float* p2Wk = (const float*)d_in[37]; const float* p2bk = (const float*)d_in[38];
    const float* p2Wv = (const float*)d_in[39]; const float* p2bv = (const float*)d_in[40];
    const float* p2Wo = (const float*)d_in[41]; const float* p2bo = (const float*)d_in[42];
    const float* p2fW = (const float*)d_in[43]; const float* p2fb = (const float*)d_in[44];
    const float* l1W = (const float*)d_in[45];  const float* l1b = (const float*)d_in[46];
    const float* l2W = (const float*)d_in[47];  const float* l2b = (const float*)d_in[48];

    // ---- workspace bump allocator (256B aligned) ----
    size_t off = 0;
    auto alloc = [&](size_t bytes) -> char* {
        char* p = (char*)d_ws + off;
        off += (bytes + 255) & ~(size_t)255;
        return p;
    };

    float* dinv  = (float*)alloc((size_t)N_NODES * 4);              // deg -> rsqrt in place
    char*  pGcn  = alloc((size_t)N_NODES * 32 * 4 * 2 + (size_t)N_NODES * 32 * 2);
    float* h32   = (float*)pGcn;                                    // [N,32] f32
    float* agg   = (float*)(pGcn + (size_t)N_NODES * 32 * 4);       // [N,32] f32
    h16*   xin16 = (h16*)  (pGcn + (size_t)N_NODES * 32 * 8);       // [N,32] f16
    char*  pHc   = alloc((size_t)N_NODES * CCH * 2);
    h16*   Hc16  = (h16*)pHc;                                       // concat [N,96] f16
    h16*   kin16 = (h16*)alloc((size_t)N_NODES * CCH * 2);          // relu(H@lin) f16
    // aliases: GCN scratch dead after layer 3; concat dead after G1
    h16*   K16   = (h16*)pGcn;                                      // [N,96] f16 (32MB region)
    h16*   V16   = (h16*)pHc;                                       // [N,96] f16

    // transposed f16 weights
    h16* wt_c2    = (h16*)alloc(32 * 32 * 2);
    h16* wt_c3    = (h16*)alloc(32 * 32 * 2);
    h16* wt_p1lin = (h16*)alloc(96 * 96 * 2);
    h16* wt_p1Wq  = (h16*)alloc(96 * 96 * 2);
    h16* wt_p1Wk  = (h16*)alloc(96 * 96 * 2);
    h16* wt_p1Wv  = (h16*)alloc(96 * 96 * 2);
    h16* wt_p1Wo  = (h16*)alloc(96 * 96 * 2);
    h16* wt_p1f   = (h16*)alloc(96 * 96 * 2);
    h16* wt_sWq   = (h16*)alloc(96 * 96 * 2);
    h16* wt_sWk   = (h16*)alloc(96 * 96 * 2);
    h16* wt_sWv   = (h16*)alloc(96 * 96 * 2);
    h16* wt_sWo   = (h16*)alloc(96 * 96 * 2);
    h16* wt_sf    = (h16*)alloc(96 * 96 * 2);
    h16* wt_p2lin = (h16*)alloc(96 * 96 * 2);
    h16* wt_p2Wq  = (h16*)alloc(96 * 96 * 2);
    h16* wt_p2Wk  = (h16*)alloc(96 * 96 * 2);
    h16* wt_p2Wv  = (h16*)alloc(96 * 96 * 2);
    h16* wt_p2Wo  = (h16*)alloc(96 * 96 * 2);
    h16* wt_p2f   = (h16*)alloc(96 * 96 * 2);
    h16* wt_l1    = (h16*)alloc(96 * 16 * 2);

    const int NZ = NGRAPH * 10 * CCH;     // 240000 (z elements)
    h16*   seed1_16 = (h16*)alloc(10 * 96 * 2);
    h16*   seed2_16 = (h16*)alloc(96 * 2);
    h16*   q1_16    = (h16*)alloc(10 * 96 * 2);
    h16*   q2_16    = (h16*)alloc(96 * 2);
    h16*   araw1    = (h16*)alloc((size_t)NZ * 2);      // attn raw, reused by SAB
    h16*   araw2    = (h16*)alloc((size_t)NGRAPH * 96 * 2);
    float* proj32   = (float*)alloc((size_t)NZ * 4);
    float* out1_32  = (float*)alloc((size_t)NZ * 4);
    h16*   out1_16  = (h16*)alloc((size_t)NZ * 2);
    float* ffn32    = (float*)alloc((size_t)NZ * 4);
    float* z32      = (float*)alloc((size_t)NZ * 4);
    h16*   z16      = (h16*)alloc((size_t)NZ * 2);
    h16*   sq16     = (h16*)alloc((size_t)NZ * 2);
    h16*   sk16     = (h16*)alloc((size_t)NZ * 2);      // reused for PMA2 K
    h16*   sv16     = (h16*)alloc((size_t)NZ * 2);      // reused for PMA2 V
    h16*   kin2_16  = (h16*)alloc((size_t)NZ * 2);
    float* proj2_32 = (float*)alloc((size_t)NGRAPH * 96 * 4);
    float* out2_32  = (float*)alloc((size_t)NGRAPH * 96 * 4);
    h16*   out2_16  = (h16*)alloc((size_t)NGRAPH * 96 * 2);
    float* ffn2_32  = (float*)alloc((size_t)NGRAPH * 96 * 4);
    float* z2_32    = (float*)alloc((size_t)NGRAPH * 96 * 4);
    h16*   z2_16    = (h16*)alloc((size_t)NGRAPH * 96 * 2);
    float* g1_32    = (float*)alloc((size_t)NGRAPH * 16 * 4);

    auto nb = [](long long n, int b) { return (unsigned)((n + b - 1) / b); };
    // Nt/K dispatch: (96,96)-><6,3>, (32,32)-><2,1>, (16,96)-><1,3>
    auto gemm = [&](const h16* A, const h16* WT, const float* bias,
                    float* o32, h16* o16, int M, int K, int Nt, int relu) {
        unsigned blocks = (unsigned)((M + 15) / 16);
        if (Nt == 96)
            k_gemm_wmma<6, 3><<<blocks, 32, 0, stream>>>(A, WT, bias, o32, o16, M, relu);
        else if (Nt == 32)
            k_gemm_wmma<2, 1><<<blocks, 32, 0, stream>>>(A, WT, bias, o32, o16, M, relu);
        else
            k_gemm_wmma<1, 3><<<blocks, 32, 0, stream>>>(A, WT, bias, o32, o16, M, relu);
        (void)K;
    };
    auto prep = [&](const float* W, h16* WT, int K, int Nt) {
        k_prep_wt<<<nb((long long)K * Nt, 256), 256, 0, stream>>>(W, WT, K, Nt);
    };

    // ---- weight prep + seed conversion ----
    prep(c2W, wt_c2, 32, 32);       prep(c3W, wt_c3, 32, 32);
    prep(p1linW, wt_p1lin, 96, 96); prep(p1Wq, wt_p1Wq, 96, 96);
    prep(p1Wk, wt_p1Wk, 96, 96);   prep(p1Wv, wt_p1Wv, 96, 96);
    prep(p1Wo, wt_p1Wo, 96, 96);   prep(p1fW, wt_p1f, 96, 96);
    prep(sWq, wt_sWq, 96, 96);     prep(sWk, wt_sWk, 96, 96);
    prep(sWv, wt_sWv, 96, 96);     prep(sWo, wt_sWo, 96, 96);
    prep(sfW, wt_sf, 96, 96);
    prep(p2linW, wt_p2lin, 96, 96); prep(p2Wq, wt_p2Wq, 96, 96);
    prep(p2Wk, wt_p2Wk, 96, 96);   prep(p2Wv, wt_p2Wv, 96, 96);
    prep(p2Wo, wt_p2Wo, 96, 96);   prep(p2fW, wt_p2f, 96, 96);
    prep(l1W, wt_l1, 96, 16);
    k_f2h<<<nb(960, 256), 256, 0, stream>>>(p1seed, seed1_16, 960);
    k_f2h<<<1, 96, 0, stream>>>(p2seed, seed2_16, 96);

    // ---- GCN normalization: deg = 1 + in-degree; dinv = rsqrt(deg) ----
    k_set1<<<nb(N_NODES, 256), 256, 0, stream>>>(dinv, N_NODES);
    k_degacc<<<nb(N_EDGES, 256), 256, 0, stream>>>(edst, dinv, N_EDGES);
    k_rsqrt_ip<<<nb(N_NODES, 256), 256, 0, stream>>>(dinv, N_NODES);

    const int NF = N_NODES * H1;
    // ---- layer 1 ----
    k_conv1<<<nb(NF, 256), 256, 0, stream>>>(x, c1W, h32);
    k_zero<<<nb(NF, 256), 256, 0, stream>>>(agg, NF);
    k_scatter<<<nb((long long)N_EDGES * 32, 256), 256, 0, stream>>>(esrc, edst, dinv, h32, agg);
    k_finalize<<<nb(NF, 256), 256, 0, stream>>>(agg, h32, dinv, c1b, xin16, Hc16, 0);
    // ---- layer 2 ----
    gemm(xin16, wt_c2, nullptr, h32, nullptr, N_NODES, 32, 32, 0);
    k_zero<<<nb(NF, 256), 256, 0, stream>>>(agg, NF);
    k_scatter<<<nb((long long)N_EDGES * 32, 256), 256, 0, stream>>>(esrc, edst, dinv, h32, agg);
    k_finalize<<<nb(NF, 256), 256, 0, stream>>>(agg, h32, dinv, c2b, xin16, Hc16, 32);
    // ---- layer 3 ----
    gemm(xin16, wt_c3, nullptr, h32, nullptr, N_NODES, 32, 32, 0);
    k_zero<<<nb(NF, 256), 256, 0, stream>>>(agg, NF);
    k_scatter<<<nb((long long)N_EDGES * 32, 256), 256, 0, stream>>>(esrc, edst, dinv, h32, agg);
    k_finalize<<<nb(NF, 256), 256, 0, stream>>>(agg, h32, dinv, c3b, xin16, Hc16, 64);

    // ---- PMA1: kin = relu(H@lin+b); K,V projections; Q from shared seed ----
    gemm(Hc16, wt_p1lin, p1linb, nullptr, kin16, N_NODES, 96, 96, 1);   // Hc dead after this
    gemm(kin16, wt_p1Wk, p1bk, nullptr, K16, N_NODES, 96, 96, 0);       // into GCN scratch
    gemm(kin16, wt_p1Wv, p1bv, nullptr, V16, N_NODES, 96, 96, 0);       // into old concat
    gemm(seed1_16, wt_p1Wq, p1bq, nullptr, q1_16, 10, 96, 96, 0);
    k_attn<<<NGRAPH, 256, NHEADS * 10 * NPER * 4, stream>>>(q1_16, 0, K16, V16, araw1, 10, NPER);
    gemm(araw1, wt_p1Wo, p1bo, proj32, nullptr, NGRAPH * 10, 96, 96, 0);
    k_addres<<<nb(NZ, 256), 256, 0, stream>>>(proj32, p1seed, 960, out1_32, out1_16, NZ);
    gemm(out1_16, wt_p1f, p1fb, ffn32, nullptr, NGRAPH * 10, 96, 96, 1);
    k_add2<<<nb(NZ, 256), 256, 0, stream>>>(out1_32, ffn32, z32, z16, NZ);

    // ---- SAB: MAB(z, z) ----
    gemm(z16, wt_sWq, sbq, nullptr, sq16, NGRAPH * 10, 96, 96, 0);
    gemm(z16, wt_sWk, sbk, nullptr, sk16, NGRAPH * 10, 96, 96, 0);
    gemm(z16, wt_sWv, sbv, nullptr, sv16, NGRAPH * 10, 96, 96, 0);
    k_attn<<<NGRAPH, 256, NHEADS * 10 * 10 * 4, stream>>>(sq16, 10 * 96, sk16, sv16, araw1, 10, 10);
    gemm(araw1, wt_sWo, sbo, proj32, nullptr, NGRAPH * 10, 96, 96, 0);
    k_add2<<<nb(NZ, 256), 256, 0, stream>>>(proj32, z32, out1_32, out1_16, NZ);
    gemm(out1_16, wt_sf, sfb, ffn32, nullptr, NGRAPH * 10, 96, 96, 1);
    k_add2<<<nb(NZ, 256), 256, 0, stream>>>(out1_32, ffn32, z32, z16, NZ);

    // ---- PMA2: single-seed pooling ----
    gemm(z16, wt_p2lin, p2linb, nullptr, kin2_16, NGRAPH * 10, 96, 96, 1);
    gemm(seed2_16, wt_p2Wq, p2bq, nullptr, q2_16, 1, 96, 96, 0);
    gemm(kin2_16, wt_p2Wk, p2bk, nullptr, sk16, NGRAPH * 10, 96, 96, 0);
    gemm(kin2_16, wt_p2Wv, p2bv, nullptr, sv16, NGRAPH * 10, 96, 96, 0);
    k_attn<<<NGRAPH, 256, NHEADS * 1 * 10 * 4, stream>>>(q2_16, 0, sk16, sv16, araw2, 1, 10);
    gemm(araw2, wt_p2Wo, p2bo, proj2_32, nullptr, NGRAPH, 96, 96, 0);
    k_addres<<<nb(NGRAPH * 96, 256), 256, 0, stream>>>(proj2_32, p2seed, 96, out2_32, out2_16, NGRAPH * 96);
    gemm(out2_16, wt_p2f, p2fb, ffn2_32, nullptr, NGRAPH, 96, 96, 1);
    k_add2<<<nb(NGRAPH * 96, 256), 256, 0, stream>>>(out2_32, ffn2_32, z2_32, z2_16, NGRAPH * 96);

    // ---- classification head ----
    gemm(z2_16, wt_l1, l1b, g1_32, nullptr, NGRAPH, 96, 16, 1);
    k_lin2<<<nb(NGRAPH * 2, 256), 256, 0, stream>>>(g1_32, l2W, l2b, (float*)d_out);
}
